// ModulatedResBlock_13288628813938
// MI455X (gfx1250) — matrix operand order repaired and checked by reading, not compiled, not run on hardware
//
#include <hip/hip_runtime.h>

// ---------------------------------------------------------------------------
// CDNA5 / gfx1250 implicit-GEMM implementation of a StyleGAN2 ModulatedResBlock.
// All matrix work goes through v_wmma_f32_16x16x32_bf16 (wave32 WMMA).
// Double-buffered LDS pipeline; A-tile fill uses GLOBAL_LOAD_ASYNC_TO_LDS_B128
// (ASYNCcnt) when the builtin is available.
// ---------------------------------------------------------------------------

typedef __attribute__((ext_vector_type(16))) __bf16 v16bf;
typedef __attribute__((ext_vector_type(8)))  float  v8f;
typedef __attribute__((ext_vector_type(4)))  int    v4i;

#define BATCH 16
#define CDIM  512
#define HWPIX 4096   // 64*64

#define AS_GLOBAL __attribute__((address_space(1)))
#define AS_LOCAL  __attribute__((address_space(3)))

#if defined(__has_builtin)
#if __has_builtin(__builtin_amdgcn_global_load_async_to_lds_b128) && \
    __has_builtin(__builtin_amdgcn_s_wait_asynccnt)
#define MRB_ASYNC 1
#endif
#endif

static __device__ __constant__ float kMOD_SCALE  = 0.044194173824159216f; // 1/sqrt(512)
static __device__ __constant__ float kCONV_SCALE = 0.014731391274719739f; // 1/sqrt(512*9)
static __device__ __constant__ float kRES_SCALE  = 0.044194173824159216f; // 1/sqrt(512)
static __device__ __constant__ float kGAIN       = 1.4142135623730951f;   // sqrt(2)
static __device__ __constant__ float kINV_SQRT2  = 0.7071067811865475f;

// -------------------------------- prep kernels -----------------------------

// s[b,c] = MOD_SCALE * dot(style[b,:], mod_w[c,:]) + mod_b[c]
__global__ void mrb_mod_kernel(const float* __restrict__ style,
                               const float* __restrict__ mod_w,
                               const float* __restrict__ mod_b,
                               float* __restrict__ s_out)
{
    int idx = blockIdx.x * blockDim.x + threadIdx.x;
    if (idx >= BATCH * CDIM) return;
    int b = idx >> 9, c = idx & 511;
    const float* st = style + (size_t)b * CDIM;
    const float* mw = mod_w + (size_t)c * CDIM;
    float acc = 0.f;
    for (int k = 0; k < CDIM; ++k) acc += st[k] * mw[k];
    s_out[idx] = acc * kMOD_SCALE + mod_b[c];
}

// Repack conv weight [o][i][3][3] f32 -> w9[k][o][i] bf16 and wsq[o][i]=sum_k w^2
__global__ void mrb_wprep_kernel(const float* __restrict__ conv_w,
                                 __bf16* __restrict__ w9,
                                 float* __restrict__ wsq)
{
    int idx = blockIdx.x * blockDim.x + threadIdx.x;
    if (idx >= CDIM * CDIM) return;
    int o = idx >> 9, i = idx & 511;
    const float* src = conv_w + (size_t)idx * 9;
    float q = 0.f;
    #pragma unroll
    for (int kk = 0; kk < 9; ++kk) {
        float v = src[kk];
        q += v * v;
        w9[((size_t)kk * CDIM + o) * CDIM + i] = (__bf16)v;
    }
    wsq[idx] = q;
}

// demod[b,o] = rsqrt(CS^2 * sum_i s[b,i]^2 * wsq[o,i] + 1e-8)
__global__ void mrb_demod_kernel(const float* __restrict__ s,
                                 const float* __restrict__ wsq,
                                 float* __restrict__ demod)
{
    int idx = blockIdx.x * blockDim.x + threadIdx.x;
    if (idx >= BATCH * CDIM) return;
    int b = idx >> 9, o = idx & 511;
    const float* sb = s + (size_t)b * CDIM;
    const float* wq = wsq + (size_t)o * CDIM;
    float acc = 0.f;
    for (int i = 0; i < CDIM; ++i) { float sv = sb[i]; acc += sv * sv * wq[i]; }
    demod[idx] = rsqrtf(acc * kCONV_SCALE * kCONV_SCALE + 1e-8f);
}

// wr[o][i] = bf16(res_w[o][i] * RES_SCALE)
__global__ void mrb_resprep_kernel(const float* __restrict__ res_w,
                                   __bf16* __restrict__ wr)
{
    int idx = blockIdx.x * blockDim.x + threadIdx.x;
    if (idx >= CDIM * CDIM) return;
    wr[idx] = (__bf16)(res_w[idx] * kRES_SCALE);
}

// xs[b,c,p] = bf16(x * s1[b,c]) ; xb[b,c,p] = bf16(x)
__global__ void mrb_premod_kernel(const float* __restrict__ x,
                                  const float* __restrict__ s1,
                                  __bf16* __restrict__ xs,
                                  __bf16* __restrict__ xb)
{
    size_t idx = (size_t)blockIdx.x * blockDim.x + threadIdx.x;
    if (idx >= (size_t)BATCH * CDIM * HWPIX) return;
    int bc = (int)(idx >> 12);  // b*512 + c
    float v = x[idx];
    xs[idx] = (__bf16)(v * s1[bc]);
    xb[idx] = (__bf16)v;
}

// ------------------------- implicit-GEMM WMMA conv -------------------------
// MODE 0: conv1 (3x3) epilogue: demod/noise/bias/lrelu*gain, write bf16 * s2 (feeds conv2)
// MODE 1: conv2 (3x3) epilogue: demod/noise/bias/lrelu*gain, write f32 to d_out
// MODE 2: residual 1x1:         d_out = (d_out + acc + bias) * 1/sqrt(2)
//
// Block: 128 couts x 64 pixels (one full image row) for one batch; 8 waves.
// Wave w owns m-tile w and all four 16-pixel n-tiles (4 accumulators).
// K loop flattened: KPOS shift positions x 16 cin-slices of 32, double-buffered
// in LDS; A-tile filled by async global->LDS b128 copies when available.
template <int MODE, int KPOS>
__global__ __launch_bounds__(256, 2)
void mrb_conv_wmma_kernel(const __bf16* __restrict__ act,
                          const __bf16* __restrict__ wgt,
                          const float*  __restrict__ demod,
                          const float*  __restrict__ s_next,
                          const float*  __restrict__ noise,
                          const float*  __restrict__ noise_w,
                          const float*  __restrict__ bias,
                          __bf16* __restrict__ z_out,
                          float*  __restrict__ f_out)
{
    // padded rows: 40 bf16 = 80 B stride (16B-aligned for b128, spreads banks)
    __shared__ __bf16 As[2][128 * 40];   // [cout_local][cin_local]
    __shared__ __bf16 Bs[2][64 * 40];    // transposed: [pixel][cin_local]

    const int t     = threadIdx.x;
    const int lane  = t & 31;
    const int wid   = t >> 5;
    const int hl    = lane >> 4;      // half-wave select
    const int lrow  = lane & 15;
    const int cout0 = blockIdx.x * 128;
    const int h     = blockIdx.y;     // image row
    const int p0    = h * 64;
    const int bz    = blockIdx.z;

    constexpr int NSTEP = KPOS * 16;  // (3x3: 144 steps, 1x1: 16 steps) of K=32

    v8f acc[4];
    #pragma unroll
    for (int nt = 0; nt < 4; ++nt)
        acc[nt] = (v8f){0.f,0.f,0.f,0.f,0.f,0.f,0.f,0.f};

    const __bf16* actb = act + (size_t)bz * CDIM * HWPIX;

    // ---- A tile fill: 128 rows x 32 cin, 512 x 16B segments, 2 per thread ----
    auto fillA = [&](int s, int buf) {
        int kk = s >> 4, c0 = (s & 15) << 5;
        #pragma unroll
        for (int it = 0; it < 2; ++it) {
            int idx = it * 256 + t;
            int row = idx >> 2, seg = idx & 3;
            const __bf16* g = wgt + ((size_t)kk * CDIM + cout0 + row) * CDIM + c0 + seg * 8;
            __bf16* l = &As[buf][row * 40 + seg * 8];
#ifdef MRB_ASYNC
            __builtin_amdgcn_global_load_async_to_lds_b128(
                (AS_GLOBAL v4i*)g, (AS_LOCAL v4i*)l, 0, 0);
#else
            *(uint4*)l = *(const uint4*)g;
#endif
        }
    };
    // ---- B tile fill: im2col of shifted input row, stored transposed ----
    auto fillB = [&](int s, int buf) {
        int kk = s >> 4, c0 = (s & 15) << 5;
        int dy = (KPOS == 9) ? (kk / 3 - 1) : 0;
        int dx = (KPOS == 9) ? (kk % 3 - 1) : 0;
        int hin = h + dy;
        bool hok = ((unsigned)hin < 64u);
        int ci = t >> 3;                         // 0..31 cin_local
        const __bf16* arow = actb + (size_t)(c0 + ci) * HWPIX + hin * 64;
        #pragma unroll
        for (int j = 0; j < 8; ++j) {
            int px  = (t & 7) * 8 + j;           // 0..63
            int win = px + dx;
            __bf16 v = (__bf16)0.f;
            if (hok && (unsigned)win < 64u) v = arow[win];
            Bs[buf][px * 40 + ci] = v;
        }
    };

    fillA(0, 0);
    fillB(0, 0);

    for (int s = 0; s < NSTEP; ++s) {
        const int buf = s & 1;
#ifdef MRB_ASYNC
        __builtin_amdgcn_s_wait_asynccnt(0);     // async fills of tile s done
#endif
        __syncthreads();                         // + ds fills of tile s visible
        if (s + 1 < NSTEP) {                     // overlap next fills with math
            fillA(s + 1, buf ^ 1);
            fillB(s + 1, buf ^ 1);
        }
        // ---- fragments per ISA 7.12.2 layouts ----
        union { v16bf v; uint4 u[2]; } ua, ub;
        // A (16x32 bf16): lane half selects K base 0/8; pairs in v0..3, +16 in v4..7
        const __bf16* ar = &As[buf][(wid * 16 + lrow) * 40];
        ua.u[0] = *(const uint4*)(ar + hl * 8);
        ua.u[1] = *(const uint4*)(ar + 16 + hl * 8);
        // B (32x16 bf16): lane half selects K base 0/8; +16 in upper 8 elements (matches A)
        #pragma unroll
        for (int nt = 0; nt < 4; ++nt) {
            const __bf16* br = &Bs[buf][(nt * 16 + lrow) * 40 + hl * 8];
            ub.u[0] = *(const uint4*)(br);
            ub.u[1] = *(const uint4*)(br + 16);
            acc[nt] = __builtin_amdgcn_wmma_f32_16x16x32_bf16(
                          false, ua.v, false, ub.v, (short)0, acc[nt], false, false);
        }
    }

    // -------------------------------- epilogue --------------------------------
    float nw = 0.f;
    if constexpr (MODE != 2) nw = noise_w[0];

    #pragma unroll
    for (int nt = 0; nt < 4; ++nt) {
        int pix = p0 + nt * 16 + lrow;
        #pragma unroll
        for (int r = 0; r < 8; ++r) {
            int o = cout0 + wid * 16 + hl * 8 + r;   // C/D layout: M = hl*8 + r
            float a = acc[nt][r];
            size_t oidx = ((size_t)bz * CDIM + o) * HWPIX + pix;
            if constexpr (MODE == 0) {
                float v = a * kCONV_SCALE * demod[bz * CDIM + o] + nw * noise[pix] + bias[o];
                v = (v > 0.f ? v : 0.2f * v) * kGAIN;
                z_out[oidx] = (__bf16)(v * s_next[bz * CDIM + o]);
            } else if constexpr (MODE == 1) {
                float v = a * kCONV_SCALE * demod[bz * CDIM + o] + nw * noise[pix] + bias[o];
                v = (v > 0.f ? v : 0.2f * v) * kGAIN;
                f_out[oidx] = v;
            } else {
                f_out[oidx] = (f_out[oidx] + a + bias[o]) * kINV_SQRT2;
            }
        }
    }
}

// ------------------------------- launcher ----------------------------------

extern "C" void kernel_launch(void* const* d_in, const int* in_sizes, int n_in,
                              void* d_out, int out_size, void* d_ws, size_t ws_size,
                              hipStream_t stream)
{
    const float* x       = (const float*)d_in[0];
    const float* style   = (const float*)d_in[1];
    const float* conv_w1 = (const float*)d_in[2];
    const float* mod_w1  = (const float*)d_in[3];
    const float* mod_b1  = (const float*)d_in[4];
    const float* nw1     = (const float*)d_in[5];
    const float* ab1     = (const float*)d_in[6];
    const float* conv_w2 = (const float*)d_in[7];
    const float* mod_w2  = (const float*)d_in[8];
    const float* mod_b2  = (const float*)d_in[9];
    const float* nw2     = (const float*)d_in[10];
    const float* ab2     = (const float*)d_in[11];
    const float* res_w   = (const float*)d_in[12];
    const float* res_b   = (const float*)d_in[13];
    const float* noise1  = (const float*)d_in[14];
    const float* noise2  = (const float*)d_in[15];
    float* out = (float*)d_out;

    char* ws = (char*)d_ws;
    size_t off = 0;
    auto alloc = [&](size_t bytes) -> char* {
        char* p = ws + off;
        off = (off + bytes + 255) & ~(size_t)255;
        return p;
    };
    float*  s1   = (float*) alloc((size_t)BATCH * CDIM * 4);
    float*  s2   = (float*) alloc((size_t)BATCH * CDIM * 4);
    float*  dm1  = (float*) alloc((size_t)BATCH * CDIM * 4);
    float*  dm2  = (float*) alloc((size_t)BATCH * CDIM * 4);
    float*  wsq1 = (float*) alloc((size_t)CDIM * CDIM * 4);
    float*  wsq2 = (float*) alloc((size_t)CDIM * CDIM * 4);
    __bf16* w1_9 = (__bf16*)alloc((size_t)9 * CDIM * CDIM * 2);
    __bf16* w2_9 = (__bf16*)alloc((size_t)9 * CDIM * CDIM * 2);
    __bf16* wr   = (__bf16*)alloc((size_t)CDIM * CDIM * 2);
    __bf16* xs1  = (__bf16*)alloc((size_t)BATCH * CDIM * HWPIX * 2);
    __bf16* xb   = (__bf16*)alloc((size_t)BATCH * CDIM * HWPIX * 2);
    __bf16* z    = (__bf16*)alloc((size_t)BATCH * CDIM * HWPIX * 2);

    // --- stage 1: modulation / weight prep ---
    mrb_mod_kernel<<<32, 256, 0, stream>>>(style, mod_w1, mod_b1, s1);
    mrb_mod_kernel<<<32, 256, 0, stream>>>(style, mod_w2, mod_b2, s2);
    mrb_wprep_kernel<<<1024, 256, 0, stream>>>(conv_w1, w1_9, wsq1);
    mrb_wprep_kernel<<<1024, 256, 0, stream>>>(conv_w2, w2_9, wsq2);
    mrb_demod_kernel<<<32, 256, 0, stream>>>(s1, wsq1, dm1);
    mrb_demod_kernel<<<32, 256, 0, stream>>>(s2, wsq2, dm2);
    mrb_resprep_kernel<<<1024, 256, 0, stream>>>(res_w, wr);
    mrb_premod_kernel<<<131072, 256, 0, stream>>>(x, s1, xs1, xb);

    // --- stage 2: WMMA implicit-GEMM convs + residual ---
    dim3 grid(4, 64, BATCH);   // cout tiles x image rows x batch
    mrb_conv_wmma_kernel<0, 9><<<grid, 256, 0, stream>>>(
        xs1, w1_9, dm1, s2, noise1, nw1, ab1, z, nullptr);
    mrb_conv_wmma_kernel<1, 9><<<grid, 256, 0, stream>>>(
        z, w2_9, dm2, nullptr, noise2, nw2, ab2, nullptr, out);
    mrb_conv_wmma_kernel<2, 1><<<grid, 256, 0, stream>>>(
        xb, wr, nullptr, nullptr, nullptr, nullptr, res_b, nullptr, out);
}
